// Attention_9921374454177
// MI455X (gfx1250) — matrix operand order repaired and checked by reading, not compiled, not run on hardware
//
#include <hip/hip_runtime.h>
#include <hip/hip_bf16.h>

typedef __bf16 bf16_t;
typedef __attribute__((ext_vector_type(8)))  float   v8f;
typedef __attribute__((ext_vector_type(8)))  __bf16  v8bf;
typedef __attribute__((ext_vector_type(16))) __bf16  v16bf;
typedef __attribute__((ext_vector_type(4)))  unsigned int v4u;
typedef __attribute__((ext_vector_type(4)))  int v4i;
typedef __attribute__((ext_vector_type(8)))  int v8i;

#define WMMA_BF16(a, b, c) \
  __builtin_amdgcn_wmma_f32_16x16x32_bf16(false, (a), false, (b), (short)0, (c), false, false)

constexpr int BATCH = 2;
constexpr int SEQ   = 2048;
constexpr int DIM   = 1024;
constexpr int HEADS = 16;
constexpr int DHEAD = 64;
constexpr int INNER = HEADS * DHEAD;   // 1024

// ---------------------------------------------------------------------------
// Fragment loaders (layouts per CDNA5 ISA §7.12.2, wave32)
// A 16x32 bf16: lane L holds M=L&15; elems 0..7 -> K = kb + (L<16?0:8)+0..7,
//               elems 8..15 -> +16.
// B 32x16 bf16 (from N-major / transposed storage): lane L holds N=L&15;
//               elems 0..7 -> K = kb + (L<16?0:8)+0..7, elems 8..15 -> +16.
// ---------------------------------------------------------------------------
__device__ inline v16bf load_a_frag(const bf16_t* base, int rowStride, int m0,
                                    int kb, int lane) {
  const int row = m0 + (lane & 15);
  const int d   = (lane < 16) ? 0 : 8;
  const bf16_t* p = base + (size_t)row * rowStride + kb + d;
  union { v16bf v; v8bf h[2]; } u;
  u.h[0] = *reinterpret_cast<const v8bf*>(p);
  u.h[1] = *reinterpret_cast<const v8bf*>(p + 16);
  return u.v;
}

__device__ inline v16bf load_b_frag(const bf16_t* baseT, int rowStride, int n0,
                                    int kb, int lane) {
  const int col = n0 + (lane & 15);
  const int d   = (lane < 16) ? 0 : 8;
  const bf16_t* p = baseT + (size_t)col * rowStride + kb + d;
  union { v16bf v; v8bf h[2]; } u;
  u.h[0] = *reinterpret_cast<const v8bf*>(p);
  u.h[1] = *reinterpret_cast<const v8bf*>(p + 16);
  return u.v;
}

// ---------------------------------------------------------------------------
// Tensor Data Mover: DMA a [rows x 64] bf16 tile (row-major, row stride 64)
// from global memory into LDS.  6-arg builtin form (clang-23 arity); D#
// packed per CDNA5 ISA (2-D tensor, groups 2/3
// and the trailing SGPR group zero).
// ---------------------------------------------------------------------------
__device__ inline void tdm_load_tile_64xN(const bf16_t* gsrc, unsigned ldsByteOff,
                                          int rows) {
  const unsigned long long ga = (unsigned long long)(size_t)gsrc;
  v4u g0;
  g0[0] = 1u;                                   // count=1, is_restore=0
  g0[1] = ldsByteOff;                           // lds_addr (bytes)
  g0[2] = (unsigned)(ga & 0xffffffffu);         // global_addr[31:0]
  g0[3] = (unsigned)((ga >> 32) & 0x1ffffffu)   // global_addr[56:32]
          | 0x80000000u;                        // type=2 ("image")
  v8i g1;
  g1[0] = 0x00010000;                           // workgroup_mask=0, data_size=1 (2B)
  g1[1] = 64 << 16;                             // tensor_dim0 = 64
  g1[2] = SEQ << 16;                            // tensor_dim1 = SEQ
  g1[3] = 64 << 16;                             // tile_dim0 = 64
  g1[4] = rows;                                 // tile_dim1 = rows
  g1[5] = 64;                                   // tensor_dim0_stride = 64
  g1[6] = 0;
  g1[7] = 0;
  const v4i gz4 = {0, 0, 0, 0};                 // groups 2/3 unused (2-D)
  const v8i gz8 = {0, 0, 0, 0, 0, 0, 0, 0};
  __builtin_amdgcn_tensor_load_to_lds(g0, g1, gz4, gz4, gz8, 0);
}

// ---------------------------------------------------------------------------
// Software-pipelined 32x64 wave tile: 8 WMMA per k-step; loads for step k+1
// are issued before the WMMAs of step k (register double-buffer), plus
// global_prefetch two k-steps ahead.
// ---------------------------------------------------------------------------
__device__ inline void gemm_tile_32x64(const bf16_t* __restrict__ A, int lda,
                                       const bf16_t* __restrict__ Bt, int ldb,
                                       int m0, int n0, int K, int lane,
                                       v8f acc[2][4]) {
  v16bf a0 = load_a_frag(A, lda, m0,      0, lane);
  v16bf a1 = load_a_frag(A, lda, m0 + 16, 0, lane);
  v16bf b0 = load_b_frag(Bt, ldb, n0,      0, lane);
  v16bf b1 = load_b_frag(Bt, ldb, n0 + 16, 0, lane);
  v16bf b2 = load_b_frag(Bt, ldb, n0 + 32, 0, lane);
  v16bf b3 = load_b_frag(Bt, ldb, n0 + 48, 0, lane);
  for (int kb = 32; kb < K; kb += 32) {
    const v16bf na0 = load_a_frag(A, lda, m0,      kb, lane);
    const v16bf na1 = load_a_frag(A, lda, m0 + 16, kb, lane);
    const v16bf nb0 = load_b_frag(Bt, ldb, n0,      kb, lane);
    const v16bf nb1 = load_b_frag(Bt, ldb, n0 + 16, kb, lane);
    const v16bf nb2 = load_b_frag(Bt, ldb, n0 + 32, kb, lane);
    const v16bf nb3 = load_b_frag(Bt, ldb, n0 + 48, kb, lane);
    if (kb + 32 < K) {   // L2 -> WGP$ prefetch two k-steps ahead
      __builtin_prefetch(A  + (size_t)(m0 + (lane & 15)) * lda + kb + 32, 0, 1);
      __builtin_prefetch(Bt + (size_t)(n0 + (lane & 15)) * ldb + kb + 32, 0, 1);
    }
    acc[0][0] = WMMA_BF16(a0, b0, acc[0][0]);
    acc[0][1] = WMMA_BF16(a0, b1, acc[0][1]);
    acc[0][2] = WMMA_BF16(a0, b2, acc[0][2]);
    acc[0][3] = WMMA_BF16(a0, b3, acc[0][3]);
    acc[1][0] = WMMA_BF16(a1, b0, acc[1][0]);
    acc[1][1] = WMMA_BF16(a1, b1, acc[1][1]);
    acc[1][2] = WMMA_BF16(a1, b2, acc[1][2]);
    acc[1][3] = WMMA_BF16(a1, b3, acc[1][3]);
    a0 = na0; a1 = na1; b0 = nb0; b1 = nb1; b2 = nb2; b3 = nb3;
  }
  acc[0][0] = WMMA_BF16(a0, b0, acc[0][0]);
  acc[0][1] = WMMA_BF16(a0, b1, acc[0][1]);
  acc[0][2] = WMMA_BF16(a0, b2, acc[0][2]);
  acc[0][3] = WMMA_BF16(a0, b3, acc[0][3]);
  acc[1][0] = WMMA_BF16(a1, b0, acc[1][0]);
  acc[1][1] = WMMA_BF16(a1, b1, acc[1][1]);
  acc[1][2] = WMMA_BF16(a1, b2, acc[1][2]);
  acc[1][3] = WMMA_BF16(a1, b3, acc[1][3]);
}

// ---------------------------------------------------------------------------
// Weight prep: wt[n][k] = (bf16) w[k][n]   (N-major for B-fragment loads)
// ---------------------------------------------------------------------------
__global__ __launch_bounds__(256) void transpose_bf16_kernel(
    const float* __restrict__ w, bf16_t* __restrict__ wt, int K, int N) {
  const size_t idx = (size_t)blockIdx.x * blockDim.x + threadIdx.x;
  if (idx >= (size_t)K * N) return;
  const int n  = (int)(idx % N);
  const int kk = (int)(idx / N);
  wt[(size_t)n * K + kk] = (bf16_t)w[idx];
}

// ---------------------------------------------------------------------------
// RMSNorm: xn = x / max(||x|| * dim^-1/2, eps) * g, cast to bf16
// ---------------------------------------------------------------------------
__global__ __launch_bounds__(256) void rmsnorm_kernel(
    const float* __restrict__ x, const float* __restrict__ g,
    bf16_t* __restrict__ xn) {
  const int row = blockIdx.x;                 // 0 .. BATCH*SEQ-1
  const float* xr = x + (size_t)row * DIM;
  float s = 0.f;
  for (int i = threadIdx.x; i < DIM; i += 256) { const float v = xr[i]; s += v * v; }
#pragma unroll
  for (int sh = 16; sh >= 1; sh >>= 1) s += __shfl_xor(s, sh, 32);
  __shared__ float sums[8];
  if ((threadIdx.x & 31) == 0) sums[threadIdx.x >> 5] = s;
  __syncthreads();
  float tot = 0.f;
#pragma unroll
  for (int w = 0; w < 8; ++w) tot += sums[w];
  const float nrm  = sqrtf(tot) * 0.03125f;   // * 1024^-0.5
  const float rinv = 1.0f / fmaxf(nrm, 1e-8f);
  bf16_t* xo = xn + (size_t)row * DIM;
  for (int i = threadIdx.x; i < DIM; i += 256)
    xo[i] = (bf16_t)(xr[i] * rinv * g[i]);
}

// ---------------------------------------------------------------------------
// QKV GEMM: [4096 x 1024] x [1024 x 3072]; wave tile = 32x64.
// Epilogue scatters into q (scaled, [b,h,n,64]), k ([b,h,n,64]), vT ([b,h,64,n]).
// ---------------------------------------------------------------------------
__global__ __launch_bounds__(256) void gemm_qkv_kernel(
    const bf16_t* __restrict__ xn, const bf16_t* __restrict__ wT,
    bf16_t* __restrict__ q, bf16_t* __restrict__ k, bf16_t* __restrict__ vT) {
  const int lane = threadIdx.x & 31;
  const int wave = (int)((blockIdx.x * blockDim.x + threadIdx.x) >> 5);
  const int TN = (3 * INNER) / 64;            // 48 col-tiles of 64
  const int tm = wave / TN, tn = wave % TN;
  const int m0 = tm * 32;

  v8f acc[2][4] = {};
  gemm_tile_32x64(xn, DIM, wT, DIM, m0, tn * 64, DIM, lane, acc);

  const int cl = lane & 15;
  const int roffL = (lane < 16) ? 0 : 8;
#pragma unroll
  for (int ms = 0; ms < 2; ++ms) {
    const int rbase = m0 + ms * 16 + roffL;
#pragma unroll
    for (int nt = 0; nt < 4; ++nt) {
      const int col = tn * 64 + nt * 16 + cl;
      const int sec = col / INNER;            // 0=q 1=k 2=v
      const int cin = col % INNER;
      const int h   = cin / DHEAD;
      const int d   = cin % DHEAD;
#pragma unroll
      for (int r = 0; r < 8; ++r) {
        const int grow = rbase + r;           // 0..4095
        const int bb = grow >> 11, nn = grow & (SEQ - 1);
        const float val = acc[ms][nt][r];
        const size_t bh = (size_t)(bb * HEADS + h);
        if (sec == 0)      q [(bh * SEQ + nn) * DHEAD + d] = (bf16_t)(val * 0.125f);
        else if (sec == 1) k [(bh * SEQ + nn) * DHEAD + d] = (bf16_t)val;
        else               vT[(bh * DHEAD + d) * SEQ + nn] = (bf16_t)val;
      }
    }
  }
}

// ---------------------------------------------------------------------------
// Online-softmax step for one 16x32 S pair (C-layout), mask templated so the
// causal compares only exist in the diagonal block.
// ---------------------------------------------------------------------------
template <bool MASK>
__device__ inline void softmax_step(const v8f& s0, const v8f& s1,
                                    float* mrow, float* lrow, v8f acc[4],
                                    bf16_t* myP, int m0, int j0, int roff, int cl) {
#pragma unroll
  for (int r = 0; r < 8; ++r) {
    float v0 = s0[r], v1 = s1[r];
    if (MASK) {
      const int row = m0 + roff + r;
      v0 = (j0 + cl      <= row) ? v0 : -3.0e38f;
      v1 = (j0 + 16 + cl <= row) ? v1 : -3.0e38f;
    }
    float mx = fmaxf(v0, v1);
#pragma unroll
    for (int sh = 8; sh >= 1; sh >>= 1) mx = fmaxf(mx, __shfl_xor(mx, sh, 32));
    const float mnew  = fmaxf(mrow[r], mx);
    const float alpha = __expf(mrow[r] - mnew);
    const float p0 = __expf(v0 - mnew);
    const float p1 = __expf(v1 - mnew);
    float rs = p0 + p1;
#pragma unroll
    for (int sh = 8; sh >= 1; sh >>= 1) rs += __shfl_xor(rs, sh, 32);
    lrow[r] = lrow[r] * alpha + rs;
    mrow[r] = mnew;
#pragma unroll
    for (int nt = 0; nt < 4; ++nt) acc[nt][r] *= alpha;
    myP[(roff + r) * 32 + cl]      = (bf16_t)p0;   // C-layout -> (row, col32)
    myP[(roff + r) * 32 + 16 + cl] = (bf16_t)p1;
  }
}

// ---------------------------------------------------------------------------
// Causal flash attention: one wave = 16 query rows of one (b,h).
// Q tile staged global->LDS by the Tensor Data Mover (async, TENSORcnt) while
// the first K fragments load; 32 keys / iteration: S = Q K^T (4 WMMA),
// online softmax, P V (4 WMMA).  K fragments double-buffered; V loads issued
// before the softmax VALU chain.  P re-laid out C->A through per-wave LDS.
// ---------------------------------------------------------------------------
__global__ __launch_bounds__(256) void attn_kernel(
    const bf16_t* __restrict__ q, const bf16_t* __restrict__ k,
    const bf16_t* __restrict__ vT, bf16_t* __restrict__ O) {
  __shared__ bf16_t pS[8][16 * 32];
  __shared__ bf16_t sQ[8][16 * 64];
  const int lane  = threadIdx.x & 31;
  const int wv    = threadIdx.x >> 5;
  const int gw    = blockIdx.x * 8 + wv;
  const int mtile = gw & (SEQ / 16 - 1);      // 128 row tiles per (b,h)
  const int bh    = gw >> 7;
  const int bb = bh >> 4, h = bh & 15;
  const int m0 = mtile * 16;

  const bf16_t* qb = q  + (size_t)bh * SEQ * DHEAD;
  const bf16_t* kp = k  + (size_t)bh * SEQ * DHEAD;
  const bf16_t* vb = vT + (size_t)bh * DHEAD * SEQ;

  // async DMA: Q tile (16 x 64 bf16) -> per-wave LDS, overlapped with K loads
  tdm_load_tile_64xN(qb + (size_t)m0 * DHEAD,
                     (unsigned)(size_t)(&sQ[wv][0]), 16);

  const int roff = (lane < 16) ? 0 : 8;
  const int cl   = lane & 15;
  float mrow[8], lrow[8];
  v8f acc[4] = {};
#pragma unroll
  for (int r = 0; r < 8; ++r) { mrow[r] = -3.0e38f; lrow[r] = 0.f; }

  bf16_t* myP = &pS[wv][0];
  const int jEnd = m0 + 16;
  // first block needing causal compares: j0+31 > m0
  const int jMaskStart = (m0 >= 31) ? ((((m0 - 31) >> 5) << 5) + 32) : 0;

  // preload K fragments for j0 = 0 (in flight while the TDM copies Q)
  v16bf kc0 = load_b_frag(kp, DHEAD, 0,  0,  lane);
  v16bf kc1 = load_b_frag(kp, DHEAD, 0,  32, lane);
  v16bf kc2 = load_b_frag(kp, DHEAD, 16, 0,  lane);
  v16bf kc3 = load_b_frag(kp, DHEAD, 16, 32, lane);

  // Q A-fragments from LDS once the tensor DMA has landed
  __builtin_amdgcn_s_wait_tensorcnt(0);
  const v16bf qa0 = load_a_frag(&sQ[wv][0], DHEAD, 0, 0,  lane);
  const v16bf qa1 = load_a_frag(&sQ[wv][0], DHEAD, 0, 32, lane);

  for (int j0 = 0; j0 < jEnd; j0 += 32) {
    v8f s0 = {}, s1 = {};
    s0 = WMMA_BF16(qa0, kc0, s0);
    s0 = WMMA_BF16(qa1, kc1, s0);
    s1 = WMMA_BF16(qa0, kc2, s1);
    s1 = WMMA_BF16(qa1, kc3, s1);

    // issue V loads (this block) + K loads (next block) before the softmax
    // VALU chain so VMEM latency hides under exp/shuffles
    const v16bf vf0 = load_b_frag(vb, SEQ, 0,  j0, lane);
    const v16bf vf1 = load_b_frag(vb, SEQ, 16, j0, lane);
    const v16bf vf2 = load_b_frag(vb, SEQ, 32, j0, lane);
    const v16bf vf3 = load_b_frag(vb, SEQ, 48, j0, lane);
    const int jn = j0 + 32;
    if (jn < jEnd) {
      kc0 = load_b_frag(kp, DHEAD, jn,      0,  lane);
      kc1 = load_b_frag(kp, DHEAD, jn,      32, lane);
      kc2 = load_b_frag(kp, DHEAD, jn + 16, 0,  lane);
      kc3 = load_b_frag(kp, DHEAD, jn + 16, 32, lane);
    }

    if (j0 >= jMaskStart)
      softmax_step<true >(s0, s1, mrow, lrow, acc, myP, m0, j0, roff, cl);
    else
      softmax_step<false>(s0, s1, mrow, lrow, acc, myP, m0, j0, roff, cl);

    // cross-lane LDS handoff: wait for DS stores before A-fragment reload
    asm volatile("s_wait_dscnt 0" ::: "memory");
    union { v16bf v; v8bf hh[2]; } pu;
    {
      const int pd = (lane < 16) ? 0 : 8;
      const bf16_t* pp = myP + cl * 32 + pd;
      pu.hh[0] = *reinterpret_cast<const v8bf*>(pp);
      pu.hh[1] = *reinterpret_cast<const v8bf*>(pp + 16);
    }
    acc[0] = WMMA_BF16(pu.v, vf0, acc[0]);
    acc[1] = WMMA_BF16(pu.v, vf1, acc[1]);
    acc[2] = WMMA_BF16(pu.v, vf2, acc[2]);
    acc[3] = WMMA_BF16(pu.v, vf3, acc[3]);
  }

#pragma unroll
  for (int r = 0; r < 8; ++r) {
    const float inv = 1.0f / lrow[r];
    const int nn = m0 + roff + r;
    const size_t orow = (size_t)bb * SEQ + nn;
#pragma unroll
    for (int nt = 0; nt < 4; ++nt)
      O[orow * INNER + h * DHEAD + nt * 16 + cl] = (bf16_t)(acc[nt][r] * inv);
  }
}

// ---------------------------------------------------------------------------
// Output GEMM: [4096 x 1024] x [1024 x 1024] -> fp32 d_out; wave tile 32x64
// ---------------------------------------------------------------------------
__global__ __launch_bounds__(256) void gemm_out_kernel(
    const bf16_t* __restrict__ O, const bf16_t* __restrict__ woutT,
    float* __restrict__ out) {
  const int lane = threadIdx.x & 31;
  const int wave = (int)((blockIdx.x * blockDim.x + threadIdx.x) >> 5);
  const int TN = DIM / 64;                    // 16 col-tiles
  const int tm = wave / TN, tn = wave % TN;
  const int m0 = tm * 32;

  v8f acc[2][4] = {};
  gemm_tile_32x64(O, INNER, woutT, INNER, m0, tn * 64, INNER, lane, acc);

  const int cl = lane & 15;
  const int roffL = (lane < 16) ? 0 : 8;
#pragma unroll
  for (int ms = 0; ms < 2; ++ms) {
    const int rbase = m0 + ms * 16 + roffL;
#pragma unroll
    for (int nt = 0; nt < 4; ++nt) {
      const int col = tn * 64 + nt * 16 + cl;
#pragma unroll
      for (int r = 0; r < 8; ++r)
        out[(size_t)(rbase + r) * DIM + col] = acc[ms][nt][r];
    }
  }
}

// ---------------------------------------------------------------------------
extern "C" void kernel_launch(void* const* d_in, const int* in_sizes, int n_in,
                              void* d_out, int out_size, void* d_ws, size_t ws_size,
                              hipStream_t stream) {
  const float* x    = (const float*)d_in[0];
  // d_in[1]: mask (all-true in this problem; causal handled in-kernel)
  const float* g    = (const float*)d_in[2];
  const float* wqkv = (const float*)d_in[3];
  const float* wout = (const float*)d_in[4];
  float* out = (float*)d_out;

  char* ws = (char*)d_ws;
  size_t off = 0;
  auto alloc = [&](size_t bytes) -> void* {
    void* p = ws + off;
    off += (bytes + 255) & ~(size_t)255;
    return p;
  };
  bf16_t* xn    = (bf16_t*)alloc((size_t)BATCH * SEQ * DIM * 2);      // reused as O
  bf16_t* wqkvT = (bf16_t*)alloc((size_t)3 * INNER * DIM * 2);
  bf16_t* woutT = (bf16_t*)alloc((size_t)DIM * INNER * 2);
  bf16_t* qd    = (bf16_t*)alloc((size_t)BATCH * HEADS * SEQ * DHEAD * 2);
  bf16_t* kd    = (bf16_t*)alloc((size_t)BATCH * HEADS * SEQ * DHEAD * 2);
  bf16_t* vTd   = (bf16_t*)alloc((size_t)BATCH * HEADS * DHEAD * SEQ * 2);
  (void)in_sizes; (void)n_in; (void)out_size; (void)ws_size;

  {
    const size_t tot = (size_t)DIM * 3 * INNER;
    transpose_bf16_kernel<<<(unsigned)((tot + 255) / 256), 256, 0, stream>>>(
        wqkv, wqkvT, DIM, 3 * INNER);
  }
  {
    const size_t tot = (size_t)INNER * DIM;
    transpose_bf16_kernel<<<(unsigned)((tot + 255) / 256), 256, 0, stream>>>(
        wout, woutT, INNER, DIM);
  }
  rmsnorm_kernel<<<BATCH * SEQ, 256, 0, stream>>>(x, g, xn);

  // 128 row-tiles (32 rows) * 48 col-tiles = 6144 waves / 8 per block
  gemm_qkv_kernel<<<(BATCH * SEQ / 32) * (3 * INNER / 64) / 8, 256, 0, stream>>>(
      xn, wqkvT, qd, kd, vTd);

  bf16_t* O = xn;  // xn dead after QKV GEMM; reuse as attention output
  attn_kernel<<<BATCH * HEADS * (SEQ / 16) / 8, 256, 0, stream>>>(qd, kd, vTd, O);

  gemm_out_kernel<<<(BATCH * SEQ / 32) * (DIM / 64) / 8, 256, 0, stream>>>(
      O, woutT, out);
}
